// WindowAttention_80685255622933
// MI455X (gfx1250) — matrix operand order repaired and checked by reading, not compiled, not run on hardware
//
#include <hip/hip_runtime.h>

// ---------------------------------------------------------------------------
// CDNA5 (gfx1250, wave32) window-attention via v_wmma_f32_16x16x32_f16
// ---------------------------------------------------------------------------

typedef _Float16 h8  __attribute__((ext_vector_type(8)));
typedef _Float16 h16 __attribute__((ext_vector_type(16)));
typedef float    f8  __attribute__((ext_vector_type(8)));

#define WMMA_F16(a, b, c) \
    __builtin_amdgcn_wmma_f32_16x16x32_f16(false, (a), false, (b), (short)0, (c), false, false)

static __device__ __forceinline__ h16 cat8(h8 lo, h8 hi) {
    return __builtin_shufflevector(lo, hi, 0,1,2,3,4,5,6,7,8,9,10,11,12,13,14,15);
}

// Problem constants
#define NB   16          // batch
#define NHW  4096        // H*W
#define NT   65536       // tokens = NB*NHW
#define NC   256         // channels
#define NH   8           // heads
#define HD   32          // head dim
#define WS2  64          // tokens per window (one image row)
#define NWIN 8192        // NB*NH*64 windows

// ---------------------------------------------------------------------------
// Prep: fp32 -> fp16 convert of x; transpose+convert weights to [out][in] f16
// ---------------------------------------------------------------------------
__global__ void cvt_x_kernel(const float* __restrict__ x, _Float16* __restrict__ xh, int n) {
    int i = blockIdx.x * blockDim.x + threadIdx.x;
    if (i < n) xh[i] = (_Float16)x[i];
}

__global__ void prep_w_kernel(const float* __restrict__ wqkv, const float* __restrict__ wproj,
                              _Float16* __restrict__ wqkvT, _Float16* __restrict__ wprojT) {
    int i = blockIdx.x * blockDim.x + threadIdx.x;      // 0 .. 768*256-1
    if (i < 768 * 256) {
        int n = i >> 8, k = i & 255;
        wqkvT[i] = (_Float16)wqkv[k * 768 + n];          // w_qkv is (in=256, out=768)
    }
    if (i < 256 * 256) {
        int n = i >> 8, k = i & 255;
        wprojT[i] = (_Float16)wproj[k * 256 + n];        // w_proj is (in=256, out=256)
    }
}

// ---------------------------------------------------------------------------
// QKV GEMM: xh[65536,256] @ wqkvT -> scatter to Q,K ([win][tok][d]) and Vt ([win][d][tok])
// One wave = 16x64 output tile. 4 waves / block.
// ---------------------------------------------------------------------------
__global__ __launch_bounds__(128) void qkv_gemm_kernel(
    const _Float16* __restrict__ xh, const _Float16* __restrict__ wT,
    const float* __restrict__ bias,
    _Float16* __restrict__ q, _Float16* __restrict__ k, _Float16* __restrict__ vt)
{
    const int lane = threadIdx.x & 31;
    const int wv   = threadIdx.x >> 5;
    const int tile = blockIdx.x * 4 + wv;     // 0 .. 4096*12-1
    const int mt   = tile / 12;
    const int ng   = tile - mt * 12;
    const int m0   = mt * 16;
    const int n0   = ng * 64;
    const int lh   = lane >> 4;               // lane half
    const int ll   = lane & 15;

    f8 acc[4];
#pragma unroll
    for (int j = 0; j < 4; ++j) {
        float bv = bias[n0 + j * 16 + ll];
#pragma unroll
        for (int r = 0; r < 8; ++r) acc[j][r] = bv;
    }

    const int arow = m0 + ll;
#pragma unroll
    for (int kk = 0; kk < 8; ++kk) {
        const int k0 = kk * 32;
        const _Float16* ap = xh + arow * NC + k0 + lh * 16;
        h16 a = *(const h16*)ap;
#pragma unroll
        for (int j = 0; j < 4; ++j) {
            const _Float16* bp = wT + (n0 + j * 16 + ll) * NC + k0 + lh * 16;
            h16 bm = *(const h16*)bp;
            acc[j] = WMMA_F16(a, bm, acc[j]);
        }
    }

    const int sec = n0 >> 8;                  // uniform: 0=Q, 1=K, 2=V
#pragma unroll
    for (int j = 0; j < 4; ++j) {
        const int n  = n0 + j * 16 + ll;
        const int cn = n & 255;
        const int h  = cn >> 5, d = cn & 31;
#pragma unroll
        for (int r = 0; r < 8; ++r) {
            const int m  = m0 + r + lh * 8;   // global token
            const int b  = m >> 12;
            const int hw = m & 4095;
            const int wr = hw >> 6, p = hw & 63;
            const int wi = (b * NH + h) * 64 + wr;
            _Float16 val = (_Float16)acc[j][r];
            if (sec == 0)      q [(wi * WS2 + p) * HD + d] = val;
            else if (sec == 1) k [(wi * WS2 + p) * HD + d] = val;
            else               vt[(wi * HD  + d) * WS2 + p] = val;
        }
    }
}

// ---------------------------------------------------------------------------
// Attention: one wave = one 16-query strip of one window (4 strips / window).
// S = Q K^T (4 WMMA), softmax (VALU + shfl), O = P V (4 WMMA, K-accum).
// ---------------------------------------------------------------------------
__global__ __launch_bounds__(128) void attn_kernel(
    const _Float16* __restrict__ q, const _Float16* __restrict__ k,
    const _Float16* __restrict__ vt, _Float16* __restrict__ o)
{
    __shared__ __align__(32) _Float16 lp[4][16 * 64];

    const int lane = threadIdx.x & 31;
    const int wv   = threadIdx.x >> 5;
    const int gid  = blockIdx.x * 4 + wv;     // 0 .. 32767
    const int wi   = gid >> 2;                // window id
    const int mt   = gid & 3;
    const int m0   = mt * 16;
    const int lh   = lane >> 4, ll = lane & 15;

    const _Float16* Qw = q  + (size_t)wi * (WS2 * HD);
    const _Float16* Kw = k  + (size_t)wi * (WS2 * HD);
    const _Float16* Vt = vt + (size_t)wi * (HD * WS2);

    // A = Q strip (16x32, K = head dim)
    const _Float16* ap = Qw + (m0 + ll) * HD + lh * 16;
    h16 a = *(const h16*)ap;

    f8 c[4];
#pragma unroll
    for (int j = 0; j < 4; ++j) {
        h16 bm = *(const h16*)(Kw + (j * 16 + ll) * HD + lh * 16);  // B[k=d][n=key]
        f8 z = {};
        c[j] = WMMA_F16(a, bm, z);
    }

    // Row softmax on the 16x64 strip (scale = 1/sqrt(32))
    const float scale = 0.17677669529663687f;
    _Float16* P = lp[wv];
#pragma unroll
    for (int r = 0; r < 8; ++r) {
        float s0 = c[0][r], s1 = c[1][r], s2 = c[2][r], s3 = c[3][r];
        float mx = fmaxf(fmaxf(s0, s1), fmaxf(s2, s3));
#pragma unroll
        for (int off = 1; off < 16; off <<= 1) mx = fmaxf(mx, __shfl_xor(mx, off, 32));
        float e0 = __expf((s0 - mx) * scale);
        float e1 = __expf((s1 - mx) * scale);
        float e2 = __expf((s2 - mx) * scale);
        float e3 = __expf((s3 - mx) * scale);
        float sm = e0 + e1 + e2 + e3;
#pragma unroll
        for (int off = 1; off < 16; off <<= 1) sm += __shfl_xor(sm, off, 32);
        float inv = 1.0f / sm;
        const int m = r + lh * 8;
        P[m * 64 +  0 + ll] = (_Float16)(e0 * inv);
        P[m * 64 + 16 + ll] = (_Float16)(e1 * inv);
        P[m * 64 + 32 + ll] = (_Float16)(e2 * inv);
        P[m * 64 + 48 + ll] = (_Float16)(e3 * inv);
    }
    __syncthreads();

    // O = P @ V : M=16, K=64 (2 steps), N=32 (2 tiles)
    f8 o0 = {}, o1 = {};
#pragma unroll
    for (int kk = 0; kk < 2; ++kk) {
        const _Float16* pp = P + ll * 64 + kk * 32 + lh * 16;
        h16 pa = *(const h16*)pp;
        h16 b0 = *(const h16*)(Vt + ( 0 + ll) * WS2 + kk * 32 + lh * 16);
        h16 b1 = *(const h16*)(Vt + (16 + ll) * WS2 + kk * 32 + lh * 16);
        o0 = WMMA_F16(pa, b0, o0);
        o1 = WMMA_F16(pa, b1, o1);
    }

    // Store to O[token][C] with C index = h*32 + d
    const int b  = wi >> 9;
    const int h  = (wi >> 6) & 7;
    const int wr = wi & 63;
    const int tb = b * NHW + wr * WS2 + m0;
#pragma unroll
    for (int r = 0; r < 8; ++r) {
        const int t = tb + r + lh * 8;
        _Float16* op = o + (size_t)t * NC + h * HD;
        op[ll]      = (_Float16)o0[r];
        op[16 + ll] = (_Float16)o1[r];
    }
}

// ---------------------------------------------------------------------------
// Output projection: O[65536,256] @ wprojT + b_proj -> out f32
// ---------------------------------------------------------------------------
__global__ __launch_bounds__(128) void proj_gemm_kernel(
    const _Float16* __restrict__ oh, const _Float16* __restrict__ wT,
    const float* __restrict__ bias, float* __restrict__ out)
{
    const int lane = threadIdx.x & 31;
    const int wv   = threadIdx.x >> 5;
    const int tile = blockIdx.x * 4 + wv;     // 0 .. 4096*4-1
    const int mt   = tile >> 2;
    const int ng   = tile & 3;
    const int m0   = mt * 16;
    const int n0   = ng * 64;
    const int lh   = lane >> 4, ll = lane & 15;

    f8 acc[4];
#pragma unroll
    for (int j = 0; j < 4; ++j) {
        float bv = bias[n0 + j * 16 + ll];
#pragma unroll
        for (int r = 0; r < 8; ++r) acc[j][r] = bv;
    }

    const int arow = m0 + ll;
#pragma unroll
    for (int kk = 0; kk < 8; ++kk) {
        const int k0 = kk * 32;
        const _Float16* ap = oh + arow * NC + k0 + lh * 16;
        h16 a = *(const h16*)ap;
#pragma unroll
        for (int j = 0; j < 4; ++j) {
            const _Float16* bp = wT + (n0 + j * 16 + ll) * NC + k0 + lh * 16;
            h16 bm = *(const h16*)bp;
            acc[j] = WMMA_F16(a, bm, acc[j]);
        }
    }

#pragma unroll
    for (int j = 0; j < 4; ++j) {
        const int n = n0 + j * 16 + ll;
#pragma unroll
        for (int r = 0; r < 8; ++r) {
            const int m = m0 + r + lh * 8;
            out[(size_t)m * NC + n] = acc[j][r];
        }
    }
}

// ---------------------------------------------------------------------------
extern "C" void kernel_launch(void* const* d_in, const int* in_sizes, int n_in,
                              void* d_out, int out_size, void* d_ws, size_t ws_size,
                              hipStream_t stream) {
    const float* x      = (const float*)d_in[0];
    const float* w_qkv  = (const float*)d_in[1];
    const float* b_qkv  = (const float*)d_in[2];
    const float* w_proj = (const float*)d_in[3];
    const float* b_proj = (const float*)d_in[4];
    float* out = (float*)d_out;

    // Workspace carve-up (all 256B aligned)
    size_t off = 0;
    auto carve = [&](size_t bytes) -> void* {
        void* p = (char*)d_ws + off;
        off += (bytes + 255) & ~(size_t)255;
        return p;
    };
    _Float16* xh     = (_Float16*)carve((size_t)NT * NC * 2);   // 32 MB
    _Float16* wqkvT  = (_Float16*)carve((size_t)768 * 256 * 2);
    _Float16* wprojT = (_Float16*)carve((size_t)256 * 256 * 2);
    _Float16* qws    = (_Float16*)carve((size_t)NT * NC * 2);   // 32 MB
    _Float16* kws    = (_Float16*)carve((size_t)NT * NC * 2);   // 32 MB
    _Float16* vtws   = (_Float16*)carve((size_t)NT * NC * 2);   // 32 MB
    _Float16* ows    = (_Float16*)carve((size_t)NT * NC * 2);   // 32 MB
    (void)ws_size;

    // 1. fp16 conversion of activations + weight transposes
    cvt_x_kernel<<<(NT * NC) / 256, 256, 0, stream>>>(x, xh, NT * NC);
    prep_w_kernel<<<(768 * 256) / 256, 256, 0, stream>>>(w_qkv, w_proj, wqkvT, wprojT);

    // 2. QKV projection: 4096 m-tiles x 12 n-groups, 4 waves/block
    qkv_gemm_kernel<<<(4096 * 12) / 4, 128, 0, stream>>>(xh, wqkvT, b_qkv, qws, kws, vtws);

    // 3. Windowed attention: 8192 windows x 4 query strips, 4 waves/block
    attn_kernel<<<(NWIN * 4) / 4, 128, 0, stream>>>(qws, kws, vtws, ows);

    // 4. Output projection: 4096 m-tiles x 4 n-groups, 4 waves/block
    proj_gemm_kernel<<<(4096 * 4) / 4, 128, 0, stream>>>(ows, wprojT, b_proj, out);

    (void)in_sizes; (void)n_in; (void)out_size;
}
